// MCGATLayer_18648747999681
// MI455X (gfx1250) — matrix operand order repaired and checked by reading, not compiled, not run on hardware
//
#include <hip/hip_runtime.h>
#include <hip/hip_bf16.h>

// ---------------------------------------------------------------------------
// Problem constants (from the reference)
// ---------------------------------------------------------------------------
#define M_USERS 20000
#define N_ITEMS 20000
#define E_EDGES 320000
#define C_CH    4
#define H_HEADS 4
#define D_IN    256
#define D_OUT   128
#define D_CAT   (C_CH * D_OUT)       // 512
#define NEG_SLOPE 0.01f

typedef __attribute__((ext_vector_type(16))) __bf16 v16bf;
typedef __attribute__((ext_vector_type(8)))  float  v8f;

// ---------------------------------------------------------------------------
// Helpers
// ---------------------------------------------------------------------------
__device__ __forceinline__ unsigned short f32_to_bf16(float f) {
    unsigned x = __float_as_uint(f);
    // round-to-nearest-even
    unsigned r = x + 0x7FFFu + ((x >> 16) & 1u);
    return (unsigned short)(r >> 16);
}

// order-preserving uint key for float atomicMax (works for all finite floats)
__device__ __forceinline__ unsigned fkey(float f) {
    unsigned u = __float_as_uint(f);
    return (u & 0x80000000u) ? ~u : (u | 0x80000000u);
}
__device__ __forceinline__ float funkey(unsigned k) {
    unsigned u = (k & 0x80000000u) ? (k & 0x7FFFFFFFu) : ~k;
    return __uint_as_float(u);
}
__device__ __forceinline__ float lrelu(float x) {
    return x > 0.f ? x : NEG_SLOPE * x;
}

// ---------------------------------------------------------------------------
// Kernels
// ---------------------------------------------------------------------------
__global__ void k_zero_u32(unsigned* p, long long n) {
    long long i = (long long)blockIdx.x * blockDim.x + threadIdx.x;
    if (i < n) p[i] = 0u;
}

__global__ void k_cvt_bf16(const float* __restrict__ in,
                           unsigned short* __restrict__ out, int n) {
    int i = blockIdx.x * blockDim.x + threadIdx.x;
    if (i < n) out[i] = f32_to_bf16(in[i]);
}

// w: (C, D_IN, D_OUT) raw params. wt: (C*D_OUT rows, D_IN cols) bf16, where
// row (c*128+n), col k holds cumsum_{cc<=c} w[cc,k,n]  (B stored N-major = K-contig)
__global__ void k_prep_W(const float* __restrict__ w,
                         unsigned short* __restrict__ wt) {
    int idx = blockIdx.x * blockDim.x + threadIdx.x;          // over 512*256
    if (idx >= D_CAT * D_IN) return;
    int nglob = idx / D_IN;                // 0..511
    int k     = idx - nglob * D_IN;        // 0..255
    int c     = nglob >> 7;                // channel
    int n     = nglob & 127;               // output dim within channel
    float acc = 0.f;
    for (int cc = 0; cc <= c; ++cc)
        acc += w[(cc * D_IN + k) * D_OUT + n];
    wt[idx] = f32_to_bf16(acc);
}

// a: (H, C, D_OUT) raw -> out: cumsum along C
__global__ void k_prep_A(const float* __restrict__ a, float* __restrict__ out) {
    int idx = blockIdx.x * blockDim.x + threadIdx.x;          // H*C*128 = 2048
    if (idx >= H_HEADS * C_CH * D_OUT) return;
    int h = idx / (C_CH * D_OUT);
    int c = (idx / D_OUT) % C_CH;
    int d = idx % D_OUT;
    float acc = 0.f;
    for (int cc = 0; cc <= c; ++cc)
        acc += a[(h * C_CH + cc) * D_OUT + d];
    out[idx] = acc;
}

// Dense GEMM:  Cout(rows x 512) = A(rows x 256, bf16) @ Bt^T
// Bt stored as (512 x 256) bf16 row-major (row = output column, contiguous K).
// One wave per 16x16 tile; v_wmma_f32_16x16x32_bf16 over K=256.
__global__ void k_gemm_wmma(const unsigned short* __restrict__ A,
                            const unsigned short* __restrict__ Bt,
                            float* __restrict__ Cout, int total_tiles) {
    int wave = threadIdx.x >> 5;                    // 0..7 (uniform per wave)
    int tile = blockIdx.x * 8 + wave;
    if (tile >= total_tiles) return;                // wave-uniform exit
    int tm = tile >> 5;                             // tiles along M (512/16=32 along N)
    int tn = tile & 31;

    int lane = threadIdx.x & 31;
    int r    = lane & 15;
    int hi   = lane >> 4;

    // A fragment source: row (tm*16+r), K chunks at hi*8 and hi*8+16 (8 bf16 each)
    const unsigned short* arow = A  + (size_t)(tm * 16 + r) * D_IN + hi * 8;
    // B fragment source: column (tn*16+r), K chunks at hi*8 and hi*8+16 (8 bf16 each)
    const unsigned short* brow = Bt + (size_t)(tn * 16 + r) * D_IN + hi * 8;

    v8f acc = {};
#pragma unroll
    for (int k0 = 0; k0 < D_IN; k0 += 32) {
        union { uint4 q[2]; v16bf v; } fa, fb;
        fa.q[0] = *(const uint4*)(arow + k0);        // K = k0+hi*8 .. +7
        fa.q[1] = *(const uint4*)(arow + k0 + 16);   // K = k0+hi*8+16 .. +23
        fb.q[0] = *(const uint4*)(brow + k0);        // K = k0+hi*8 .. +7
        fb.q[1] = *(const uint4*)(brow + k0 + 16);   // K = k0+hi*8+16 .. +23
        acc = __builtin_amdgcn_wmma_f32_16x16x32_bf16(
                  false, fa.v, false, fb.v, (short)0, acc, false, false);
    }

    // C/D layout: VGPR v -> row (v + 8*hi), col = lane&15
    float* cbase = Cout + (size_t)(tm * 16 + hi * 8) * D_CAT + tn * 16 + r;
#pragma unroll
    for (int v = 0; v < 8; ++v) cbase[(size_t)v * D_CAT] = acc[v];
}

// node-level attention scores: s[node*16 + c*4 + h] = sum_d h[node,c*128+d]*Acs[h,c,d]
__global__ void k_scores(const float* __restrict__ h,
                         const float* __restrict__ Acs,
                         float* __restrict__ s, int nnodes) {
    int idx = blockIdx.x * blockDim.x + threadIdx.x;
    if (idx >= nnodes * 16) return;
    int node = idx >> 4;
    int ch   = idx & 15;
    int c    = ch >> 2;
    int hh   = ch & 3;
    const float* hp = h + (size_t)node * D_CAT + c * D_OUT;
    const float* ap = Acs + (hh * C_CH + c) * D_OUT;
    float acc = 0.f;
#pragma unroll 4
    for (int d = 0; d < D_OUT; ++d) acc += hp[d] * ap[d];
    s[(size_t)node * 16 + c * 4 + hh] = acc;
}

// pass 1: segment max over destination row (uint-keyed atomicMax), all heads
__global__ void k_att_max(const int* __restrict__ row, const int* __restrict__ col,
                          const float* __restrict__ ssrc, const float* __restrict__ sdst,
                          unsigned* __restrict__ mx) {
    int e = blockIdx.x * blockDim.x + threadIdx.x;
    if (e >= E_EDGES) return;
    int c = blockIdx.y;
    int rr = row[(size_t)c * E_EDGES + e];
    int cc = col[(size_t)c * E_EDGES + e];
    float4 av = *(const float4*)(ssrc + (size_t)rr * 16 + c * 4);
    float4 bv = *(const float4*)(sdst + (size_t)cc * 16 + c * 4);
    float as[4] = {av.x, av.y, av.z, av.w};
    float bs[4] = {bv.x, bv.y, bv.z, bv.w};
#pragma unroll
    for (int h = 0; h < H_HEADS; ++h) {
        float lg = lrelu(as[h] + bs[h]);
        atomicMax(&mx[(size_t)rr * 16 + c * 4 + h], fkey(lg));
    }
}

// pass 2: segment sum of exp(logit - max)
__global__ void k_att_den(const int* __restrict__ row, const int* __restrict__ col,
                          const float* __restrict__ ssrc, const float* __restrict__ sdst,
                          const unsigned* __restrict__ mx, float* __restrict__ den) {
    int e = blockIdx.x * blockDim.x + threadIdx.x;
    if (e >= E_EDGES) return;
    int c = blockIdx.y;
    int rr = row[(size_t)c * E_EDGES + e];
    int cc = col[(size_t)c * E_EDGES + e];
    float4 av = *(const float4*)(ssrc + (size_t)rr * 16 + c * 4);
    float4 bv = *(const float4*)(sdst + (size_t)cc * 16 + c * 4);
    float as[4] = {av.x, av.y, av.z, av.w};
    float bs[4] = {bv.x, bv.y, bv.z, bv.w};
#pragma unroll
    for (int h = 0; h < H_HEADS; ++h) {
        float lg = lrelu(as[h] + bs[h]);
        float m  = funkey(mx[(size_t)rr * 16 + c * 4 + h]);
        atomicAdd(&den[(size_t)rr * 16 + c * 4 + h], __expf(lg - m));
    }
}

// pass 3: per-edge combined weight w_e = sum_h exp(lg-mx)/den  (heads share h_dst)
__global__ void k_att_weight(const int* __restrict__ row, const int* __restrict__ col,
                             const float* __restrict__ ssrc, const float* __restrict__ sdst,
                             const unsigned* __restrict__ mx, const float* __restrict__ den,
                             float* __restrict__ wbuf) {
    int e = blockIdx.x * blockDim.x + threadIdx.x;
    if (e >= E_EDGES) return;
    int c = blockIdx.y;
    int rr = row[(size_t)c * E_EDGES + e];
    int cc = col[(size_t)c * E_EDGES + e];
    float4 av = *(const float4*)(ssrc + (size_t)rr * 16 + c * 4);
    float4 bv = *(const float4*)(sdst + (size_t)cc * 16 + c * 4);
    float as[4] = {av.x, av.y, av.z, av.w};
    float bs[4] = {bv.x, bv.y, bv.z, bv.w};
    float w = 0.f;
#pragma unroll
    for (int h = 0; h < H_HEADS; ++h) {
        float lg = lrelu(as[h] + bs[h]);
        float m  = funkey(mx[(size_t)rr * 16 + c * 4 + h]);
        float d  = den[(size_t)rr * 16 + c * 4 + h];
        w += __expf(lg - m) / d;
    }
    wbuf[(size_t)c * E_EDGES + e] = w;
}

// pass 4: scatter  out[row, c*128 + d] += w_e * h_dst[col, c*128 + d]
// one wave per edge, 4 dims per lane
__global__ void k_att_scatter(const int* __restrict__ row, const int* __restrict__ col,
                              const float* __restrict__ wbuf,
                              const float* __restrict__ hdst,
                              float* __restrict__ outp) {
    int wave = threadIdx.x >> 5;
    int e = blockIdx.x * 8 + wave;
    if (e >= E_EDGES) return;
    int c = blockIdx.y;
    int lane = threadIdx.x & 31;
    int rr = row[(size_t)c * E_EDGES + e];
    int cc = col[(size_t)c * E_EDGES + e];
    float w = wbuf[(size_t)c * E_EDGES + e];
    float4 hv = *(const float4*)(hdst + (size_t)cc * D_CAT + c * D_OUT + lane * 4);
    float* ob = outp + (size_t)rr * D_CAT + c * D_OUT + lane * 4;
    atomicAdd(ob + 0, w * hv.x);
    atomicAdd(ob + 1, w * hv.y);
    atomicAdd(ob + 2, w * hv.z);
    atomicAdd(ob + 3, w * hv.w);
}

__global__ void k_relu(float* p, long long n) {
    long long i = (long long)blockIdx.x * blockDim.x + threadIdx.x;
    if (i < n) { float v = p[i]; p[i] = v > 0.f ? v : 0.f; }
}

// ---------------------------------------------------------------------------
// Launcher
// ---------------------------------------------------------------------------
extern "C" void kernel_launch(void* const* d_in, const int* in_sizes, int n_in,
                              void* d_out, int out_size, void* d_ws, size_t ws_size,
                              hipStream_t stream) {
    const float* u_prev  = (const float*)d_in[0];
    const float* i_prev  = (const float*)d_in[1];
    const float* w_user  = (const float*)d_in[2];
    const float* w_item  = (const float*)d_in[3];
    const float* a_u_src = (const float*)d_in[4];
    const float* a_u_dst = (const float*)d_in[5];
    const float* a_i_src = (const float*)d_in[6];
    const float* a_i_dst = (const float*)d_in[7];
    const int* u2i_row   = (const int*)d_in[8];
    const int* u2i_col   = (const int*)d_in[9];
    const int* i2u_row   = (const int*)d_in[10];
    const int* i2u_col   = (const int*)d_in[11];
    float* out = (float*)d_out;

    // ---- workspace carve-up (256B aligned regions) ----
    char* base = (char*)d_ws;
    size_t off = 0;
    auto carve = [&](size_t bytes) -> char* {
        char* p = base + off;
        off = (off + bytes + 255) & ~(size_t)255;
        return p;
    };
    unsigned short* ubf   = (unsigned short*)carve((size_t)M_USERS * D_IN * 2);
    unsigned short* ibf   = (unsigned short*)carve((size_t)N_ITEMS * D_IN * 2);
    unsigned short* wtU   = (unsigned short*)carve((size_t)D_CAT * D_IN * 2);
    unsigned short* wtI   = (unsigned short*)carve((size_t)D_CAT * D_IN * 2);
    float* A_us = (float*)carve((size_t)H_HEADS * C_CH * D_OUT * 4);
    float* A_ud = (float*)carve((size_t)H_HEADS * C_CH * D_OUT * 4);
    float* A_is = (float*)carve((size_t)H_HEADS * C_CH * D_OUT * 4);
    float* A_id = (float*)carve((size_t)H_HEADS * C_CH * D_OUT * 4);
    float* h_u  = (float*)carve((size_t)M_USERS * D_CAT * 4);
    float* h_i  = (float*)carve((size_t)N_ITEMS * D_CAT * 4);
    float* s_u_src = (float*)carve((size_t)M_USERS * 16 * 4);
    float* s_u_dst = (float*)carve((size_t)M_USERS * 16 * 4);
    float* s_i_src = (float*)carve((size_t)N_ITEMS * 16 * 4);
    float* s_i_dst = (float*)carve((size_t)N_ITEMS * 16 * 4);
    unsigned* mx_u = (unsigned*)carve((size_t)M_USERS * 16 * 4);
    unsigned* mx_i = (unsigned*)carve((size_t)N_ITEMS * 16 * 4);
    float* den_u   = (float*)carve((size_t)M_USERS * 16 * 4);
    float* den_i   = (float*)carve((size_t)N_ITEMS * 16 * 4);
    float* w_u2i   = (float*)carve((size_t)C_CH * E_EDGES * 4);
    float* w_i2u   = (float*)carve((size_t)C_CH * E_EDGES * 4);
    (void)ws_size; (void)in_sizes; (void)n_in; (void)out_size;

    const int T = 256;
    auto blocks = [](long long n, int t) { return (unsigned)((n + t - 1) / t); };

    // ---- 0) zero accumulators + output ----
    long long out_n = (long long)(M_USERS + N_ITEMS) * D_CAT;
    k_zero_u32<<<blocks(out_n, T), T, 0, stream>>>((unsigned*)out, out_n);
    k_zero_u32<<<blocks((long long)M_USERS * 16, T), T, 0, stream>>>(mx_u, (long long)M_USERS * 16);
    k_zero_u32<<<blocks((long long)N_ITEMS * 16, T), T, 0, stream>>>(mx_i, (long long)N_ITEMS * 16);
    k_zero_u32<<<blocks((long long)M_USERS * 16, T), T, 0, stream>>>((unsigned*)den_u, (long long)M_USERS * 16);
    k_zero_u32<<<blocks((long long)N_ITEMS * 16, T), T, 0, stream>>>((unsigned*)den_i, (long long)N_ITEMS * 16);

    // ---- 1) bf16 conversions + weight/attention-vector prep (cumsum fused) ----
    k_cvt_bf16<<<blocks((long long)M_USERS * D_IN, T), T, 0, stream>>>(u_prev, ubf, M_USERS * D_IN);
    k_cvt_bf16<<<blocks((long long)N_ITEMS * D_IN, T), T, 0, stream>>>(i_prev, ibf, N_ITEMS * D_IN);
    k_prep_W<<<blocks((long long)D_CAT * D_IN, T), T, 0, stream>>>(w_user, wtU);
    k_prep_W<<<blocks((long long)D_CAT * D_IN, T), T, 0, stream>>>(w_item, wtI);
    unsigned ablk = blocks(H_HEADS * C_CH * D_OUT, T);
    k_prep_A<<<ablk, T, 0, stream>>>(a_u_src, A_us);
    k_prep_A<<<ablk, T, 0, stream>>>(a_u_dst, A_ud);
    k_prep_A<<<ablk, T, 0, stream>>>(a_i_src, A_is);
    k_prep_A<<<ablk, T, 0, stream>>>(a_i_dst, A_id);

    // ---- 2) WMMA GEMMs: h = x @ W_cat  (all 4 channels at once) ----
    int tiles = (M_USERS / 16) * (D_CAT / 16);   // 1250 * 32 = 40000
    k_gemm_wmma<<<(tiles + 7) / 8, 256, 0, stream>>>(ubf, wtU, h_u, tiles);
    k_gemm_wmma<<<(tiles + 7) / 8, 256, 0, stream>>>(ibf, wtI, h_i, tiles);

    // ---- 3) node-level attention scores ----
    k_scores<<<blocks((long long)M_USERS * 16, T), T, 0, stream>>>(h_u, A_us, s_u_src, M_USERS);
    k_scores<<<blocks((long long)N_ITEMS * 16, T), T, 0, stream>>>(h_i, A_ud, s_i_dst, N_ITEMS);
    k_scores<<<blocks((long long)N_ITEMS * 16, T), T, 0, stream>>>(h_i, A_is, s_i_src, N_ITEMS);
    k_scores<<<blocks((long long)M_USERS * 16, T), T, 0, stream>>>(h_u, A_id, s_u_dst, M_USERS);

    // ---- 4) segment softmax + scatter, both directions, all channels ----
    dim3 eg(blocks(E_EDGES, T), C_CH);
    // u2i: rows are users, dst features are items
    k_att_max<<<eg, T, 0, stream>>>(u2i_row, u2i_col, s_u_src, s_i_dst, mx_u);
    k_att_max<<<eg, T, 0, stream>>>(i2u_row, i2u_col, s_i_src, s_u_dst, mx_i);
    k_att_den<<<eg, T, 0, stream>>>(u2i_row, u2i_col, s_u_src, s_i_dst, mx_u, den_u);
    k_att_den<<<eg, T, 0, stream>>>(i2u_row, i2u_col, s_i_src, s_u_dst, mx_i, den_i);
    k_att_weight<<<eg, T, 0, stream>>>(u2i_row, u2i_col, s_u_src, s_i_dst, mx_u, den_u, w_u2i);
    k_att_weight<<<eg, T, 0, stream>>>(i2u_row, i2u_col, s_i_src, s_u_dst, mx_i, den_i, w_i2u);

    dim3 sg((E_EDGES + 7) / 8, C_CH);
    k_att_scatter<<<sg, 256, 0, stream>>>(u2i_row, u2i_col, w_u2i, h_i, out);
    k_att_scatter<<<sg, 256, 0, stream>>>(i2u_row, i2u_col, w_i2u, h_u, out + (size_t)M_USERS * D_CAT);

    // ---- 5) final ReLU in place ----
    k_relu<<<blocks(out_n, T), T, 0, stream>>>(out, out_n);
}
